// _IPEXAttention_14860586844722
// MI455X (gfx1250) — matrix operand order repaired and checked by reading, not compiled, not run on hardware
//
#include <hip/hip_runtime.h>
#include <hip/hip_bf16.h>

// Decode-phase paged attention (GQA) for MI455X / gfx1250.
// Memory-bound (~2 FLOP/byte @ 23.3 TB/s HBM): stream K/V exactly once,
// math via v_wmma_f32_16x16x32_bf16 (bf16 in, fp32 accumulate).
//   B=64 batch, H=32 q-heads, KVH=8 kv-heads (G=4), HD=128, BLOCK=64, MAXB=32.
// One workgroup per (batch, kv_head); 8 wave32s split the KV length and merge
// flash-attention partials in LDS. Next tile (contiguous 16KB K + 16KB V) is
// prefetched with global_prefetch_b8 while the current tile's WMMAs run.

#define B_     64
#define H_     32
#define KVH_   8
#define HD_    128
#define G_     4
#define BLOCK_ 64
#define MAXB_  32
#define NWAVES 8

typedef __attribute__((ext_vector_type(16))) __bf16 v16bf;
typedef __attribute__((ext_vector_type(8)))  float  v8f;
typedef __attribute__((ext_vector_type(4)))  float  v4f;

__global__ __launch_bounds__(256)
void paged_attn_decode_wmma(const float* __restrict__ q,
                            const float* __restrict__ kc,
                            const float* __restrict__ vc,
                            const int*   __restrict__ bt,
                            const int*   __restrict__ lens,
                            float*       __restrict__ out)
{
    __shared__ float LO[NWAVES][G_][HD_];   // unnormalized per-wave outputs
    __shared__ float Lm[NWAVES][G_];        // per-wave running max
    __shared__ float Ll[NWAVES][G_];        // per-wave running sum

    const int b     = blockIdx.x / KVH_;
    const int kh    = blockIdx.x % KVH_;
    const int tid   = threadIdx.x;
    const int wave  = tid >> 5;
    const int lane  = tid & 31;
    const int l16   = lane & 15;
    const int lhalf = lane >> 4;
    const int kbA   = lhalf * 8;    // A-matrix / D-matrix kv base within 16
    const int kbB   = lhalf * 16;   // unused; B operands use kbA (table order)
    const int len   = lens[b];
    const float scale = 0.088388347648318447f;  // 1/sqrt(128)

    // ---- Q into B-matrix layout (N=head, K=dim), bf16, pre-scaled ----------
    // lane holds column N=l16 (heads >= G_ zero-filled), K = 32c + kbA + e (+8 for e>=8).
    v16bf qb[4];
    {
        const bool hv = (l16 < G_);
        const float* qrow = q + ((size_t)b * H_ + (size_t)kh * G_ + (hv ? l16 : 0)) * HD_;
        #pragma unroll
        for (int c = 0; c < 4; ++c) {
            float t[16];
            *(v4f*)&t[0]  = *(const v4f*)(qrow + 32*c + kbA + 0);
            *(v4f*)&t[4]  = *(const v4f*)(qrow + 32*c + kbA + 4);
            *(v4f*)&t[8]  = *(const v4f*)(qrow + 32*c + kbA + 16);
            *(v4f*)&t[12] = *(const v4f*)(qrow + 32*c + kbA + 20);
            #pragma unroll
            for (int e = 0; e < 16; ++e)
                qb[c][e] = hv ? (__bf16)(t[e] * scale) : (__bf16)0.0f;
        }
    }

    const v8f vzero = {};
    v8f acc[8];                                   // O accum: M=head, N=16c+l16
    #pragma unroll
    for (int c = 0; c < 8; ++c) acc[c] = vzero;
    float m_run = -1e30f;
    float l_run = 0.0f;

    // ---- flash loop over this wave's 32-position KV tiles ------------------
    for (int kv0 = wave * 32; kv0 < len; kv0 += NWAVES * 32) {
        const int blk = bt[b * MAXB_ + (kv0 >> 6)];          // tile fits in one block
        const size_t tbase = (((size_t)blk * KVH_ + kh) * BLOCK_ + (size_t)(kv0 & 63)) * HD_;
        const float* kp = kc + tbase;
        const float* vp = vc + tbase;

        // Prefetch next tile (32 rows x 512B, contiguous): 4 instructions per
        // cache cover K and V fully (32 lanes x 128B-spaced lines each).
        const int kvn = kv0 + NWAVES * 32;                   // wave-uniform branch
        if (kvn < len) {
            const int blkn = bt[b * MAXB_ + (kvn >> 6)];
            const size_t nbase = (((size_t)blkn * KVH_ + kh) * BLOCK_ + (size_t)(kvn & 63)) * HD_;
            const float* kn = kc + nbase;
            const float* vn = vc + nbase;
            #pragma unroll
            for (int j = 0; j < 4; ++j) {
                __builtin_prefetch(kn + (size_t)(j * 32 + lane) * 32, 0, 3);
                __builtin_prefetch(vn + (size_t)(j * 32 + lane) * 32, 0, 3);
            }
        }

        // S^T = K_tile x Q^T : two 16(kv) x 16(head) f32 subtiles
        v8f s0 = vzero, s1 = vzero;
        #pragma unroll
        for (int c = 0; c < 4; ++c) {
            const float* r0 = kp + (size_t)l16 * HD_ + 32*c + kbA;   // kv row l16
            const float* r1 = r0 + (size_t)16 * HD_;                 // kv row l16+16
            v16bf a0, a1;
            {
                float t[16];
                *(v4f*)&t[0]  = *(const v4f*)(r0 + 0);
                *(v4f*)&t[4]  = *(const v4f*)(r0 + 4);
                *(v4f*)&t[8]  = *(const v4f*)(r0 + 16);
                *(v4f*)&t[12] = *(const v4f*)(r0 + 20);
                #pragma unroll
                for (int e = 0; e < 16; ++e) a0[e] = (__bf16)t[e];
            }
            {
                float t[16];
                *(v4f*)&t[0]  = *(const v4f*)(r1 + 0);
                *(v4f*)&t[4]  = *(const v4f*)(r1 + 4);
                *(v4f*)&t[8]  = *(const v4f*)(r1 + 16);
                *(v4f*)&t[12] = *(const v4f*)(r1 + 20);
                #pragma unroll
                for (int e = 0; e < 16; ++e) a1[e] = (__bf16)t[e];
            }
            s0 = __builtin_amdgcn_wmma_f32_16x16x32_bf16(false, a0, false, qb[c],
                                                         (short)0, s0, false, false);
            s1 = __builtin_amdgcn_wmma_f32_16x16x32_bf16(false, a1, false, qb[c],
                                                         (short)0, s1, false, false);
        }

        // mask + online softmax (head = l16; kv rows kbA+r and 16+kbA+r)
        float mt = -1e30f;
        #pragma unroll
        for (int r = 0; r < 8; ++r) {
            if (kv0 + kbA + r      >= len) s0[r] = -1e30f;
            if (kv0 + 16 + kbA + r >= len) s1[r] = -1e30f;
            mt = fmaxf(mt, fmaxf(s0[r], s1[r]));
        }
        mt = fmaxf(mt, __shfl_xor(mt, 16, 32));           // other kv half, same head
        const float mn    = fmaxf(m_run, mt);
        const float alpha = __expf(m_run - mn);
        float lt = 0.0f;
        float p0[8], p1[8];
        #pragma unroll
        for (int r = 0; r < 8; ++r) {
            p0[r] = __expf(s0[r] - mn);
            p1[r] = __expf(s1[r] - mn);
            lt += p0[r] + p1[r];
        }
        lt += __shfl_xor(lt, 16, 32);
        l_run = l_run * alpha + lt;
        m_run = mn;

        // pack P directly into A-matrix layout (M=head, K=kv) for P·V
        v16bf pa;
        #pragma unroll
        for (int r = 0; r < 8; ++r) { pa[r] = (__bf16)p0[r]; pa[8+r] = (__bf16)p1[r]; }

        // rescale accumulator rows by per-head alpha (rows>=4 are padding)
        #pragma unroll
        for (int r = 0; r < 8; ++r) {
            const float ar = __shfl(alpha, r, 32);
            #pragma unroll
            for (int c = 0; c < 8; ++c) acc[c][r] *= ar;
        }

        // O += P x V_tile : B-matrix = V (K=kv=kbA+e (+8 for e>=8), N=dim=16c+l16)
        #pragma unroll
        for (int c = 0; c < 8; ++c) {
            v16bf vbm;
            #pragma unroll
            for (int e = 0; e < 16; ++e)
                vbm[e] = (__bf16)vp[(size_t)(kbA + e + (e >= 8 ? 8 : 0)) * HD_ + 16*c + l16];
            acc[c] = __builtin_amdgcn_wmma_f32_16x16x32_bf16(false, pa, false, vbm,
                                                             (short)0, acc[c], false, false);
        }
    }

    // ---- per-wave partials to LDS ------------------------------------------
    if (lane < G_) { Lm[wave][lane] = m_run; Ll[wave][lane] = l_run; }
    if (lhalf == 0) {
        #pragma unroll
        for (int r = 0; r < G_; ++r)        // row M=r == head r (lanes 0-15)
            #pragma unroll
            for (int c = 0; c < 8; ++c)
                LO[wave][r][16*c + l16] = acc[c][r];
    }
    __syncthreads();

    // ---- merge waves (split-K flash combine), write output -----------------
    #pragma unroll
    for (int i = 0; i < (G_ * HD_) / 256; ++i) {
        const int idx = tid + i * 256;
        const int h = idx >> 7;
        const int d = idx & (HD_ - 1);
        float M = Lm[0][h];
        #pragma unroll
        for (int w = 1; w < NWAVES; ++w) M = fmaxf(M, Lm[w][h]);
        float den = 0.0f, num = 0.0f;
        #pragma unroll
        for (int w = 0; w < NWAVES; ++w) {
            const float wgt = __expf(Lm[w][h] - M);
            den += wgt * Ll[w][h];
            num += wgt * LO[w][h][d];
        }
        out[((size_t)b * H_ + (size_t)kh * G_ + h) * HD_ + d] = num / den;
    }
}

extern "C" void kernel_launch(void* const* d_in, const int* in_sizes, int n_in,
                              void* d_out, int out_size, void* d_ws, size_t ws_size,
                              hipStream_t stream) {
    (void)in_sizes; (void)n_in; (void)out_size; (void)d_ws; (void)ws_size;
    const float* q  = (const float*)d_in[0];
    const float* kc = (const float*)d_in[1];
    const float* vc = (const float*)d_in[2];
    const int*   bt = (const int*)d_in[3];
    const int*   ln = (const int*)d_in[4];
    float* out = (float*)d_out;

    dim3 grid(B_ * KVH_);   // 512 workgroups: one per (batch, kv_head)
    dim3 block(256);        // 8 wave32s split the KV length
    paged_attn_decode_wmma<<<grid, block, 0, stream>>>(q, kc, vc, bt, ln, out);
}
